// Zconv_81913616270171
// MI455X (gfx1250) — matrix operand order repaired and checked by reading, not compiled, not run on hardware
//
#include <hip/hip_runtime.h>
#include <hip/hip_bf16.h>

// ---------------------------------------------------------------------------
// MI455X (gfx1250) pipeline:
//   K1: point conv(8->64)+BN+ReLU + gather + f32 atomic scatter into voxel sums
//   K2: voxel mean -> bf16 scatter into dense src[U][32][64] + pillar counts
//   K3/K4: one-time bf16 transposes of w1/w2 to [N][K] (LDS-tiled for w1)
//   K5: GEMM1 [U,2048]x[2048,1024] bf16 WMMA, BN+ReLU, bf16 out (compute bound)
//   K6: GEMM2 [U,1024]x[1024,64]  bf16 WMMA, BN+ReLU, occupancy mask, f32 out
// GEMM tile refills use GLOBAL_LOAD_ASYNC_TO_LDS_B128 (ASYNCcnt) when the
// toolchain exposes the builtins, else global_load_b128 + ds_store_b128.
// ---------------------------------------------------------------------------

typedef __attribute__((ext_vector_type(16))) __bf16 v16bf;
typedef __attribute__((ext_vector_type(8)))  __bf16 v8bf;
typedef __attribute__((ext_vector_type(8)))  float  v8f;
typedef __attribute__((ext_vector_type(4)))  int    v4i;

#define EPS 1e-3f
#define LDK 40   // padded LDS row stride in bf16 (32 data + 8 pad => 80B rows)

#if defined(__has_builtin)
#  if __has_builtin(__builtin_amdgcn_global_load_async_to_lds_b128) && \
      __has_builtin(__builtin_amdgcn_s_wait_asynccnt)
#    define USE_ASYNC_LDS 1
#  endif
#endif
#ifndef USE_ASYNC_LDS
#  define USE_ASYNC_LDS 0
#endif

#if USE_ASYNC_LDS
typedef __attribute__((address_space(1))) v4i gas_v4i;
typedef __attribute__((address_space(3))) v4i las_v4i;

__device__ __forceinline__ void async_cp16(__bf16* lds_dst, const __bf16* gsrc)
{
    __builtin_amdgcn_global_load_async_to_lds_b128(
        (gas_v4i*)gsrc, (las_v4i*)lds_dst, 0, 0);
}
#endif

// ---------------------------- stage 1: points ------------------------------
__global__ __launch_bounds__(256) void point_scatter_kernel(
    const float* __restrict__ points, const float* __restrict__ sparse_feat,
    const int* __restrict__ inv_pillar, const int* __restrict__ unq_inv_vox,
    const float* __restrict__ w_conv,
    const float* __restrict__ g0, const float* __restrict__ b0,
    const float* __restrict__ m0, const float* __restrict__ v0,
    float* __restrict__ vsum, float* __restrict__ vcnt, int N)
{
    int gid = blockIdx.x * 256 + threadIdx.x;
    int i = gid >> 6, c = gid & 63;
    if (i >= N) return;
    const float* p = points + (size_t)i * 9 + 1;
    float dot = 0.f;
#pragma unroll
    for (int k = 0; k < 8; ++k) dot += p[k] * w_conv[k * 64 + c];
    float sc = g0[c] * rsqrtf(v0[c] + EPS);
    float val = (dot - m0[c]) * sc + b0[c];
    val = fmaxf(val, 0.f);
    val += sparse_feat[(size_t)inv_pillar[i] * 64 + c];
    int v = unq_inv_vox[i];
    atomicAdd(&vsum[(size_t)v * 64 + c], val);
    if (c == 0) atomicAdd(&vcnt[v], 1.f);
}

// ---------------------------- stage 2: voxels ------------------------------
__global__ __launch_bounds__(256) void voxel_to_bins_kernel(
    const float* __restrict__ vsum, const float* __restrict__ vcnt,
    const int* __restrict__ zbin, const int* __restrict__ pillar_inv,
    __bf16* __restrict__ srcb, int* __restrict__ pcnt, int V)
{
    int gid = blockIdx.x * 256 + threadIdx.x;
    int v = gid >> 6, c = gid & 63;
    if (v >= V) return;
    float mean = vsum[(size_t)v * 64 + c] / fmaxf(vcnt[v], 1.f);
    int u = pillar_inv[v], z = zbin[v];
    srcb[((size_t)u * 32 + z) * 64 + c] = (__bf16)mean;
    if (c == 0) atomicAdd(&pcnt[u], 1);
}

// ---------------- w1 transpose: 32x32 LDS tiles, coalesced both ways -------
__global__ __launch_bounds__(256) void transpose_w1_kernel(
    const float* __restrict__ w1, __bf16* __restrict__ w1t)
{
    // w1: [2048][1024] f32 -> w1t: [1024][2048] bf16
    __shared__ __bf16 tile[32][33];
    const int kt = blockIdx.x;             // 64 tiles along K
    const int nt = blockIdx.y;             // 32 tiles along N
    const int tx = threadIdx.x & 31;
    const int ty0 = threadIdx.x >> 5;      // 8 rows, loop x4
#pragma unroll
    for (int s = 0; s < 4; ++s) {
        int ty = ty0 + s * 8;
        tile[ty][tx] = (__bf16)w1[(size_t)(kt * 32 + ty) * 1024 + nt * 32 + tx];
    }
    __syncthreads();
#pragma unroll
    for (int s = 0; s < 4; ++s) {
        int ty = ty0 + s * 8;
        w1t[(size_t)(nt * 32 + ty) * 2048 + kt * 32 + tx] = tile[tx][ty];
    }
}

__global__ __launch_bounds__(256) void transpose_w2_kernel(
    const float* __restrict__ w2, __bf16* __restrict__ w2t)
{
    int gid = blockIdx.x * 256 + threadIdx.x;      // 64*1024
    int c = gid >> 10, k = gid & 1023;             // w2: [1024][64]
    w2t[gid] = (__bf16)w2[(size_t)k * 64 + c];
}

// ----------------------------- WMMA helpers --------------------------------
__device__ __forceinline__ v16bf frag_cat(const __bf16* lo, const __bf16* hi)
{
    v8bf a = *(const v8bf*)lo;
    v8bf b = *(const v8bf*)hi;
    return __builtin_shufflevector(a, b, 0, 1, 2, 3, 4, 5, 6, 7,
                                         8, 9, 10, 11, 12, 13, 14, 15);
}

// --------------------- GEMM1: src[U,2048] x w1t -> h[U,1024] ---------------
__global__ __launch_bounds__(256) void gemm1_wmma_kernel(
    const __bf16* __restrict__ A,    // [U][2048]
    const __bf16* __restrict__ Bt,   // [1024][2048]
    const float* __restrict__ g1, const float* __restrict__ b1,
    const float* __restrict__ m1, const float* __restrict__ v1,
    __bf16* __restrict__ Hb, int U)
{
    __shared__ __bf16 As[2][128 * LDK];
    __shared__ __bf16 Bs[2][128 * LDK];

    const int t = threadIdx.x;
    const int lane = t & 31, wave = t >> 5;
    const int bm0 = blockIdx.x * 128;
    const int bn0 = blockIdx.y * 128;
    const int wm = (wave >> 2) * 64;   // 2 wave-rows
    const int wn = (wave & 3) * 32;    // 4 wave-cols
    const int l15 = lane & 15, half = lane >> 4;

    const int lrow = t >> 1;           // 0..127
    const int lk = (t & 1) * 16;       // 0 / 16

    // clamped A row: unconditional loads, OOB rows discarded in epilogue
    int arow = bm0 + lrow;
    if (arow > U - 1) arow = U - 1;
    const __bf16* Arow = A + (size_t)arow * 2048 + lk;
    const __bf16* Brow = Bt + (size_t)(bn0 + lrow) * 2048 + lk;

    v8f acc[4][2] = {};

#if USE_ASYNC_LDS
    auto async_tile = [&](int kk, int buf) {
        __bf16* la = &As[buf][lrow * LDK + lk];
        __bf16* lb = &Bs[buf][lrow * LDK + lk];
        async_cp16(la,     Arow + kk);
        async_cp16(la + 8, Arow + kk + 8);
        async_cp16(lb,     Brow + kk);
        async_cp16(lb + 8, Brow + kk + 8);
    };
    async_tile(0, 0);
    __builtin_amdgcn_s_wait_asynccnt(0);
    __syncthreads();
#else
    uint4 ra0, ra1, rb0, rb1;
    auto loadAB = [&](int kk) {
        const uint4* pa = (const uint4*)(Arow + kk);
        ra0 = pa[0]; ra1 = pa[1];
        const uint4* pb = (const uint4*)(Brow + kk);
        rb0 = pb[0]; rb1 = pb[1];
    };
    auto stash = [&](int buf) {
        uint4* da = (uint4*)(&As[buf][lrow * LDK + lk]);
        da[0] = ra0; da[1] = ra1;
        uint4* db = (uint4*)(&Bs[buf][lrow * LDK + lk]);
        db[0] = rb0; db[1] = rb1;
    };
    loadAB(0);
    stash(0);
    __syncthreads();
#endif

    for (int it = 0; it < 64; ++it) {
        const int buf = it & 1;
        const bool more = (it + 1) < 64;
#if USE_ASYNC_LDS
        if (more) async_tile((it + 1) * 32, buf ^ 1);
#else
        if (more) loadAB((it + 1) * 32);
#endif
        if (it + 4 < 64) __builtin_prefetch(Arow + (it + 4) * 32, 0, 1);

        v16bf af[4], bfr[2];
#pragma unroll
        for (int i = 0; i < 4; ++i) {
            const __bf16* pa = &As[buf][(wm + i * 16 + l15) * LDK + half * 8];
            af[i] = frag_cat(pa, pa + 16);   // K: h*8+0..7 | h*8+16..23
        }
#pragma unroll
        for (int j = 0; j < 2; ++j) {
            const __bf16* pb = &Bs[buf][(wn + j * 16 + l15) * LDK + half * 8];
            bfr[j] = frag_cat(pb, pb + 16);  // K: h*8+0..7 | h*8+16..23
        }
#pragma unroll
        for (int i = 0; i < 4; ++i)
#pragma unroll
            for (int j = 0; j < 2; ++j)
                acc[i][j] = __builtin_amdgcn_wmma_f32_16x16x32_bf16(
                    false, af[i], false, bfr[j], (short)0, acc[i][j], false, false);

#if USE_ASYNC_LDS
        if (more) __builtin_amdgcn_s_wait_asynccnt(0);
#else
        if (more) stash(buf ^ 1);
#endif
        __syncthreads();
    }

#pragma unroll
    for (int j = 0; j < 2; ++j) {
        int col = bn0 + wn + j * 16 + l15;
        float sc = g1[col] * rsqrtf(v1[col] + EPS);
        float sh = b1[col] - m1[col] * sc;
#pragma unroll
        for (int i = 0; i < 4; ++i) {
            int rbase = bm0 + wm + i * 16 + half * 8;
#pragma unroll
            for (int r = 0; r < 8; ++r) {
                int row = rbase + r;
                if (row < U) {
                    float x = fmaxf(acc[i][j][r] * sc + sh, 0.f);
                    Hb[(size_t)row * 1024 + col] = (__bf16)x;
                }
            }
        }
    }
}

// ------------------ GEMM2: h[U,1024] x w2t -> out[U,64] --------------------
__global__ __launch_bounds__(256) void gemm2_wmma_kernel(
    const __bf16* __restrict__ A,    // [U][1024]
    const __bf16* __restrict__ Bt,   // [64][1024]
    const float* __restrict__ g2, const float* __restrict__ b2,
    const float* __restrict__ m2, const float* __restrict__ v2,
    const int* __restrict__ pcnt,
    float* __restrict__ out, int U)
{
    __shared__ __bf16 As[2][128 * LDK];
    __shared__ __bf16 Bs[2][64 * LDK];

    const int t = threadIdx.x;
    const int lane = t & 31, wave = t >> 5;
    const int bm0 = blockIdx.x * 128;
    const int wm = (wave >> 2) * 64;   // 2 wave-rows of 64
    const int wn = (wave & 3) * 16;    // 4 wave-cols of 16 (N=64 total)
    const int l15 = lane & 15, half = lane >> 4;

    const int lrow = t >> 1;           // A: 0..127
    const int lk = (t & 1) * 16;
    const int brow = t >> 2;           // B: 0..63
    const int bk = (t & 3) * 8;

    int arow = bm0 + lrow;
    if (arow > U - 1) arow = U - 1;
    const __bf16* Arow = A + (size_t)arow * 1024 + lk;
    const __bf16* Brow = Bt + (size_t)brow * 1024 + bk;

    v8f acc[4] = {};

#if USE_ASYNC_LDS
    auto async_tile = [&](int kk, int buf) {
        __bf16* la = &As[buf][lrow * LDK + lk];
        async_cp16(la,     Arow + kk);
        async_cp16(la + 8, Arow + kk + 8);
        async_cp16(&Bs[buf][brow * LDK + bk], Brow + kk);
    };
    async_tile(0, 0);
    __builtin_amdgcn_s_wait_asynccnt(0);
    __syncthreads();
#else
    uint4 ra0, ra1, rb0;
    auto loadAB = [&](int kk) {
        const uint4* pa = (const uint4*)(Arow + kk);
        ra0 = pa[0]; ra1 = pa[1];
        rb0 = *(const uint4*)(Brow + kk);
    };
    auto stash = [&](int buf) {
        uint4* da = (uint4*)(&As[buf][lrow * LDK + lk]);
        da[0] = ra0; da[1] = ra1;
        *(uint4*)(&Bs[buf][brow * LDK + bk]) = rb0;
    };
    loadAB(0);
    stash(0);
    __syncthreads();
#endif

    for (int it = 0; it < 32; ++it) {
        const int buf = it & 1;
        const bool more = (it + 1) < 32;
#if USE_ASYNC_LDS
        if (more) async_tile((it + 1) * 32, buf ^ 1);
#else
        if (more) loadAB((it + 1) * 32);
#endif
        if (it + 4 < 32) __builtin_prefetch(Arow + (it + 4) * 32, 0, 1);

        v16bf af[4], bfr;
#pragma unroll
        for (int i = 0; i < 4; ++i) {
            const __bf16* pa = &As[buf][(wm + i * 16 + l15) * LDK + half * 8];
            af[i] = frag_cat(pa, pa + 16);
        }
        {
            const __bf16* pb = &Bs[buf][(wn + l15) * LDK + half * 8];
            bfr = frag_cat(pb, pb + 16);
        }
#pragma unroll
        for (int i = 0; i < 4; ++i)
            acc[i] = __builtin_amdgcn_wmma_f32_16x16x32_bf16(
                false, af[i], false, bfr, (short)0, acc[i], false, false);

#if USE_ASYNC_LDS
        if (more) __builtin_amdgcn_s_wait_asynccnt(0);
#else
        if (more) stash(buf ^ 1);
#endif
        __syncthreads();
    }

    {
        int col = wn + l15;  // 0..63
        float sc = g2[col] * rsqrtf(v2[col] + EPS);
        float sh = b2[col] - m2[col] * sc;
#pragma unroll
        for (int i = 0; i < 4; ++i) {
            int rbase = bm0 + wm + i * 16 + half * 8;
#pragma unroll
            for (int r = 0; r < 8; ++r) {
                int row = rbase + r;
                if (row < U) {
                    float x = fmaxf(acc[i][r] * sc + sh, 0.f);
                    bool occ = pcnt[row] >= 2;
                    out[(size_t)row * 64 + col] = occ ? x : 0.f;
                }
            }
        }
    }
}

// --------------------------------- host ------------------------------------
extern "C" void kernel_launch(void* const* d_in, const int* in_sizes, int n_in,
                              void* d_out, int out_size, void* d_ws, size_t ws_size,
                              hipStream_t stream)
{
    const float* points      = (const float*)d_in[0];
    const float* sparse_feat = (const float*)d_in[1];
    const int*   inv_pillar  = (const int*)d_in[2];
    const int*   unq_inv_vox = (const int*)d_in[3];
    const int*   zbin        = (const int*)d_in[4];
    const int*   pillar_inv  = (const int*)d_in[5];
    const float* w_conv      = (const float*)d_in[6];
    const float* g0 = (const float*)d_in[7];
    const float* b0 = (const float*)d_in[8];
    const float* m0 = (const float*)d_in[9];
    const float* v0 = (const float*)d_in[10];
    const float* w1 = (const float*)d_in[11];
    const float* g1 = (const float*)d_in[12];
    const float* b1 = (const float*)d_in[13];
    const float* m1 = (const float*)d_in[14];
    const float* v1 = (const float*)d_in[15];
    const float* w2 = (const float*)d_in[16];
    const float* g2 = (const float*)d_in[17];
    const float* b2 = (const float*)d_in[18];
    const float* m2 = (const float*)d_in[19];
    const float* v2 = (const float*)d_in[20];

    const int N = in_sizes[0] / 9;   // points
    const int V = in_sizes[4];       // zbin has one entry per voxel
    const int U = out_size / 64;     // pillars

    char* ws = (char*)d_ws;
    size_t off = 0;
    auto carve = [&](size_t bytes) -> char* {
        char* p = ws + off;
        off += (bytes + 255) & ~(size_t)255;
        return p;
    };
    float*  vsum = (float*)carve((size_t)V * 64 * 4);
    float*  vcnt = (float*)carve((size_t)V * 4);
    int*    pcnt = (int*)carve((size_t)U * 4);
    __bf16* srcb = (__bf16*)carve((size_t)U * 2048 * 2);
    __bf16* hbuf = (__bf16*)carve((size_t)U * 1024 * 2);
    __bf16* w1t  = (__bf16*)carve((size_t)2048 * 1024 * 2);
    __bf16* w2t  = (__bf16*)carve((size_t)1024 * 64 * 2);
    (void)ws_size; (void)n_in;

    (void)hipMemsetAsync(vsum, 0, (size_t)V * 64 * 4, stream);
    (void)hipMemsetAsync(vcnt, 0, (size_t)V * 4, stream);
    (void)hipMemsetAsync(pcnt, 0, (size_t)U * 4, stream);
    (void)hipMemsetAsync(srcb, 0, (size_t)U * 2048 * 2, stream);

    {
        dim3 grid(64, 32);  // 2048/32 x 1024/32
        transpose_w1_kernel<<<grid, 256, 0, stream>>>(w1, w1t);
    }
    transpose_w2_kernel<<<(64 * 1024) / 256, 256, 0, stream>>>(w2, w2t);

    {
        long long tot = (long long)N * 64;
        int blocks = (int)((tot + 255) / 256);
        point_scatter_kernel<<<blocks, 256, 0, stream>>>(
            points, sparse_feat, inv_pillar, unq_inv_vox, w_conv,
            g0, b0, m0, v0, vsum, vcnt, N);
    }
    {
        long long tot = (long long)V * 64;
        int blocks = (int)((tot + 255) / 256);
        voxel_to_bins_kernel<<<blocks, 256, 0, stream>>>(
            vsum, vcnt, zbin, pillar_inv, srcb, pcnt, V);
    }
    {
        dim3 grid((U + 127) / 128, 8);
        gemm1_wmma_kernel<<<grid, 256, 0, stream>>>(
            srcb, w1t, g1, b1, m1, v1, hbuf, U);
    }
    {
        dim3 grid((U + 127) / 128, 1);
        gemm2_wmma_kernel<<<grid, 256, 0, stream>>>(
            hbuf, w2t, g2, b2, m2, v2, pcnt, (float*)d_out, U);
    }
}
